// PRPN_67345087201507
// MI455X (gfx1250) — matrix operand order repaired and checked by reading, not compiled, not run on hardware
//
#include <hip/hip_runtime.h>
#include <cstdint>
#include <cstddef>

// ---------------------------------------------------------------------------
// PRPN forward for MI455X (gfx1250, wave32).
//   * all GEMMs on v_wmma_f32_16x16x32_bf16 (fp32 accumulate)
//   * A tiles: global_load_async_to_lds_b128 + s_wait_asynccnt (double buffer)
//   * W tiles: Tensor Data Mover (tensor_load_to_lds, 6-arg builtin) +
//     s_wait_tensorcnt, one 128x64 bf16 tile (16KB) per K-chunk, double buffered
//   * ~44MB bf16 weight set resident in 192MB L2 across 256 recurrent steps
// ---------------------------------------------------------------------------

#define T_DIM 256
#define B_DIM 64
#define H_DIM 1024
#define NSLOT 15
#define NLAY  2
#define KTOT  2048            // every GEMM here is K=2048 (two 1024 halves)
#define KCH   64              // K elements staged per chunk (2 WMMAs)
#define NCH   (KTOT / KCH)    // 32 chunks

typedef __attribute__((ext_vector_type(16))) __bf16 v16bf;
typedef __attribute__((ext_vector_type(8)))  __bf16 v8bf;
typedef __attribute__((ext_vector_type(8)))  float  v8f;
typedef __attribute__((ext_vector_type(4)))  unsigned int v4u;
typedef __attribute__((ext_vector_type(8)))  int v8i;
typedef __attribute__((ext_vector_type(4)))  int v4i;

union F16x32 { v16bf v; v8bf h[2]; };

__device__ __forceinline__ float sigf(float x) { return 1.0f / (1.0f + __expf(-x)); }

__device__ __forceinline__ void async_ld16(void* lds, const void* g) {
  uint32_t l  = (uint32_t)(uint64_t)lds;   // low 32b of LDS-aperture addr = byte offset
  uint64_t ga = (uint64_t)g;
  asm volatile("global_load_async_to_lds_b128 %0, %1, off" :: "v"(l), "v"(ga) : "memory");
}

// TDM: load a (rows=128) x (K=64) bf16 tile (row pitch KTOT elems) into LDS.
__device__ __forceinline__ void tdm_load_w(const __bf16* gsrc, const void* lds) {
  uint64_t ga = (uint64_t)gsrc;
  uint32_t lo = (uint32_t)(uint64_t)lds;
  v4u g0 = { 0x1u,                                   // count=1, normal mode
             lo,                                     // lds_addr [63:32]
             (uint32_t)ga,                           // global_addr [95:64]
             (uint32_t)((ga >> 32) & 0x01FFFFFFu) | 0x80000000u }; // [120:96] + type=2
  v8i g1 = { (int)0x00010000,                        // data_size=2B
             (int)0xFFFF0000,                        // bar_addr=0 | tensor_dim0 lo16
             (int)0xFFFF7FFF,                        // tensor_dim0 hi16 | tensor_dim1 lo16
             (int)(0x7FFF | (KCH << 16)),            // tensor_dim1 hi16 | tile_dim0=64
             (int)128,                               // tile_dim1=128 rows, tile_dim2=0
             (int)KTOT,                              // tensor_dim0_stride = 2048 elems
             0, 0 };
  v4i g2 = {0, 0, 0, 0};
  v4i g3 = {0, 0, 0, 0};
  v8i g4 = {0, 0, 0, 0, 0, 0, 0, 0};
  __builtin_amdgcn_tensor_load_to_lds(g0, g1, g2, g3, g4, 0);
}

// C[M,N] = epilogue( [A0|A1](bf16, M x 2048) @ W(bf16, N x 2048)^T + bias )
// epilogue: 0 -> fp32 C = v + bias
//           1 -> bf16 C = relu(gamma*(v+bias)+beta)
//           2 -> fp32 C = tanh(gamma*(v+bias)+beta)
// grid = (N/128, M/16), block = 256 (8 waves, one 16x16 N-tile per wave)
__global__ __launch_bounds__(256)
void gemm_bf16_wmma(const __bf16* __restrict__ A0, const __bf16* __restrict__ A1,
                    const __bf16* __restrict__ W,
                    const float* __restrict__ bias,
                    const float* __restrict__ gamma,
                    const float* __restrict__ beta,
                    float* __restrict__ Cf, __bf16* __restrict__ Cb,
                    const __bf16* __restrict__ zbuf,
                    int N, int zero_rows, int epilogue)
{
  const int tid  = threadIdx.x;
  const int lane = tid & 31;
  const int wave = tid >> 5;
  const int m0   = blockIdx.y << 4;
  const int nblk = blockIdx.x << 7;
  const int n0   = nblk + (wave << 4);

  __shared__ __align__(16) __bf16 sA[2][16 * KCH];    // 2 x 2KB
  __shared__ __align__(16) __bf16 sW[2][128 * KCH];   // 2 x 16KB

  // A staging (tid < 128): 16 rows x 8 segments of 16B; bases hoisted out of loop
  const int lrow = tid >> 3;
  const int seg  = tid & 7;
  const int r    = m0 + lrow;
  const __bf16* s0 =
      ((zero_rows && r < zero_rows) ? zbuf : (A0 + (size_t)r * H_DIM)) + (seg << 3);
  const __bf16* s1 = A1 + (size_t)r * H_DIM + (seg << 3);
  __bf16* aDst0 = &sA[0][lrow * KCH + (seg << 3)];
  __bf16* aDst1 = &sA[1][lrow * KCH + (seg << 3)];

  const __bf16* Wblk = W + (size_t)nblk * KTOT;

  if (tid < 128) async_ld16(aDst0, s0);
  if (wave == 0) tdm_load_w(Wblk, &sW[0][0]);

  v8f acc = {};
  const int fr = lane & 15;
  const int kb = (lane >> 4) << 3;   // A: K blocks {kb..kb+7, kb+16..kb+23}
  const int kw = (lane >> 4) << 3;   // W: K blocks {kw..kw+7, kw+16..kw+23}
  const int wrow = (wave << 4) + fr;

  for (int c = 0; c < NCH; ++c) {
    const int buf = c & 1;
    if (c + 1 < NCH) {
      const int cn = c + 1;
      const __bf16* asrc = (cn < NCH / 2) ? (s0 + (cn << 6)) : (s1 + ((cn - NCH / 2) << 6));
      if (tid < 128) async_ld16(buf ? aDst0 : aDst1, asrc);
      if (wave == 0) tdm_load_w(Wblk + (cn << 6), &sW[buf ^ 1][0]);
      asm volatile("s_wait_asynccnt 0x1" ::: "memory");
      if (wave == 0) __builtin_amdgcn_s_wait_tensorcnt(1);
    } else {
      asm volatile("s_wait_asynccnt 0x0" ::: "memory");
      if (wave == 0) __builtin_amdgcn_s_wait_tensorcnt(0);
    }
    __syncthreads();

#pragma unroll
    for (int u = 0; u < 2; ++u) {
      F16x32 a, b;
      const __bf16* ap = &sA[buf][fr * KCH + (u << 5) + kb];
      a.h[0] = *(const v8bf*)ap;
      a.h[1] = *(const v8bf*)(ap + 16);
      const __bf16* wp = &sW[buf][wrow * KCH + (u << 5) + kw];
      b.h[0] = *(const v8bf*)wp;
      b.h[1] = *(const v8bf*)(wp + 16);
      acc = __builtin_amdgcn_wmma_f32_16x16x32_bf16(false, a.v, false, b.v,
                                                    (short)0, acc, false, false);
    }
    __syncthreads();
  }

  const int rr  = (lane >> 4) << 3;
  const int col = n0 + (lane & 15);
  const float bc = bias  ? bias[col]  : 0.0f;
  const float gc = gamma ? gamma[col] : 1.0f;
  const float tc = beta  ? beta[col]  : 0.0f;
#pragma unroll
  for (int i = 0; i < 8; ++i) {
    const int row = m0 + i + rr;
    float v = acc[i] + bc;
    if (epilogue == 1) {
      v = fmaxf(0.0f, gc * v + tc);
      Cb[(size_t)row * N + col] = (__bf16)v;
    } else if (epilogue == 2) {
      v = tanhf(gc * v + tc);
      Cf[(size_t)row * N + col] = v;
    } else {
      Cf[(size_t)row * N + col] = v;
    }
  }
}

// ---------------------------------------------------------------------------
__global__ __launch_bounds__(256)
void embed_kernel(const int* __restrict__ tok, const float* __restrict__ tab,
                  __bf16* __restrict__ out)
{
  size_t idx = (size_t)blockIdx.x * 256 + threadIdx.x;
  if (idx >= (size_t)T_DIM * B_DIM * H_DIM) return;
  size_t r = idx / H_DIM, i = idx % H_DIM;
  out[idx] = (__bf16)tab[(size_t)tok[r] * H_DIM + i];
}

__global__ __launch_bounds__(256)
void cvt_bf16_kernel(const float* __restrict__ src, __bf16* __restrict__ dst, long n)
{
  long idx = (long)blockIdx.x * 256 + threadIdx.x;
  if (idx < n) dst[idx] = (__bf16)src[idx];
}

// conv1_w (H, H, 2) -> (o, [k0 half | k1 half]) bf16
__global__ __launch_bounds__(256)
void pack_conv1_kernel(const float* __restrict__ w, __bf16* __restrict__ wp)
{
  long idx = (long)blockIdx.x * 256 + threadIdx.x;
  if (idx >= (long)H_DIM * H_DIM) return;
  long o = idx / H_DIM, i = idx % H_DIM;
  wp[o * KTOT + i]         = (__bf16)w[(o * H_DIM + i) * 2 + 0];
  wp[o * KTOT + H_DIM + i] = (__bf16)w[(o * H_DIM + i) * 2 + 1];
}

// wih|whh -> (j, o, [wih row | whh row]) ; bpack = bih + bhh
__global__ __launch_bounds__(256)
void pack_gates_kernel(const float* __restrict__ wih, const float* __restrict__ whh,
                       __bf16* __restrict__ wp)
{
  long idx = (long)blockIdx.x * 256 + threadIdx.x;
  if (idx >= (long)NLAY * 4 * H_DIM * H_DIM) return;
  long row = idx / H_DIM;            // j*4H + o
  long i   = idx % H_DIM;
  wp[row * KTOT + i]         = (__bf16)wih[idx];
  wp[row * KTOT + H_DIM + i] = (__bf16)whh[idx];
}

__global__ __launch_bounds__(256)
void pack_bias_kernel(const float* __restrict__ bih, const float* __restrict__ bhh,
                      float* __restrict__ bp)
{
  int idx = blockIdx.x * 256 + threadIdx.x;
  if (idx < NLAY * 4 * H_DIM) bp[idx] = bih[idx] + bhh[idx];
}

// gate[b,t] = sigmoid(dot(w2[g], h1[row, g*512:...]) + b2[g]); row = t*B + b
__global__ __launch_bounds__(256)
void conv2_kernel(const __bf16* __restrict__ h1, const float* __restrict__ w2,
                  const float* __restrict__ b2,
                  float* __restrict__ gate, float* __restrict__ gaten)
{
  int r  = blockIdx.x;
  int tid = threadIdx.x;
  int g  = tid >> 7, lt = tid & 127;
  const __bf16* hr = h1 + (size_t)r * H_DIM + g * 512;
  float p = 0.0f;
  for (int i = lt; i < 512; i += 128) p += (float)hr[i] * w2[g * 512 + i];
  for (int off = 16; off > 0; off >>= 1) p += __shfl_down(p, off);
  __shared__ float acc[2];
  if (tid < 2) acc[tid] = 0.0f;
  __syncthreads();
  if ((tid & 31) == 0) atomicAdd(&acc[g], p);
  __syncthreads();
  if (tid == 0) {
    int b = r % B_DIM, t = r / B_DIM;
    gate[(size_t)b * T_DIM + t]  = sigf(acc[0] + b2[0]);
    gaten[(size_t)b * T_DIM + t] = sigf(acc[1] + b2[1]);
  }
}

__global__ __launch_bounds__(256)
void mg_kernel(const float* __restrict__ gate, const float* __restrict__ gaten,
               float* __restrict__ mg, float* __restrict__ mgn)
{
  int idx = blockIdx.x * 256 + threadIdx.x;
  if (idx >= B_DIM * T_DIM) return;
  int b = idx / T_DIM, t = idx % T_DIM;
  float g = gate[idx], gn = gaten[idx];
  float cm = 1.0f, cmn = 1.0f;
  size_t base = ((size_t)b * T_DIM + t) * NSLOT;
  for (int s = 0; s < NSLOT; ++s) {
    float gh = (t - s >= 0) ? gate[(size_t)b * T_DIM + (t - s)] : 1.0e8f;
    cm  *= sigf((g  - gh) * 100.0f + 5.0f);
    cmn *= sigf((gn - gh) * 100.0f + 5.0f);
    mg[base + s]  = cm;
    mgn[base + s] = cmn;
  }
}

// one block per batch element; ring slot s -> phys (t+s)%15
__global__ __launch_bounds__(128)
void attn_kernel(float* __restrict__ mh, const float* __restrict__ mc,
                 const float* __restrict__ key, const float* __restrict__ mask,
                 int t, __bf16* __restrict__ sel_bf, float* __restrict__ selc,
                 const float* __restrict__ ring_src, __bf16* __restrict__ last_bf)
{
  int b = blockIdx.x, tid = threadIdx.x;
  __shared__ float slog[NSLOT];
  if (tid < NSLOT) slog[tid] = 0.0f;
  __syncthreads();
  const float* kv = key + (size_t)b * H_DIM;
  for (int s = 0; s < NSLOT; ++s) {
    int phys = (t + s) % NSLOT;
    const float* row = mh + ((size_t)b * NSLOT + phys) * H_DIM;
    float p = 0.0f;
    for (int i = tid; i < H_DIM; i += 128) p += row[i] * kv[i];
    for (int off = 16; off > 0; off >>= 1) p += __shfl_down(p, off);
    if ((tid & 31) == 0) atomicAdd(&slog[s], p);
  }
  __syncthreads();
  const float* mrow = mask + ((size_t)b * T_DIM + t) * NSLOT;
  float att[NSLOT], mx = -1.0e30f, sum = 0.0f;
  for (int s = 0; s < NSLOT; ++s) { att[s] = slog[s] * 0.03125f; mx = fmaxf(mx, att[s]); }
  for (int s = 0; s < NSLOT; ++s) { att[s] = __expf(att[s] - mx) * mrow[s]; sum += att[s]; }
  float inv = 1.0f / (sum + 1e-8f);
  for (int s = 0; s < NSLOT; ++s) att[s] *= inv;
  for (int i = tid; i < H_DIM; i += 128) {
    float ah = 0.0f, ac = 0.0f;
    for (int s = 0; s < NSLOT; ++s) {
      int phys = (t + s) % NSLOT;
      size_t o = ((size_t)b * NSLOT + phys) * H_DIM + i;
      ah += att[s] * mh[o];
      if (mc) ac += att[s] * mc[o];
    }
    sel_bf[(size_t)b * H_DIM + i] = (__bf16)ah;
    if (selc) selc[(size_t)b * H_DIM + i] = ac;
  }
  if (ring_src) {        // predictor mode: append h into the pm ring after use
    __syncthreads();
    int pw = t % NSLOT;
    float* dst = mh + ((size_t)b * NSLOT + pw) * H_DIM;
    for (int i = tid; i < H_DIM; i += 128) {
      float hv = ring_src[(size_t)b * H_DIM + i];
      dst[i] = hv;
      last_bf[(size_t)b * H_DIM + i] = (__bf16)hv;
    }
  }
}

__global__ __launch_bounds__(256)
void lstm_kernel(const float* __restrict__ gates, const float* __restrict__ selc,
                 float* __restrict__ rh, float* __restrict__ rc, int t,
                 __bf16* __restrict__ hlast_bf, __bf16* __restrict__ hcur_bf,
                 float* __restrict__ hcur_f, __bf16* __restrict__ hs_bf)
{
  int idx = blockIdx.x * 256 + threadIdx.x;
  if (idx >= B_DIM * H_DIM) return;
  int b = idx / H_DIM, i = idx % H_DIM;
  size_t gb = (size_t)b * 4 * H_DIM;
  float ig = gates[gb + i], fg = gates[gb + H_DIM + i];
  float gg = gates[gb + 2 * H_DIM + i], og = gates[gb + 3 * H_DIM + i];
  float c  = sigf(fg) * selc[idx] + sigf(ig) * tanhf(gg);
  float h  = sigf(og) * tanhf(c);
  int phys = t % NSLOT;
  size_t o = ((size_t)b * NSLOT + phys) * H_DIM + i;
  rh[o] = h; rc[o] = c;
  hlast_bf[idx] = (__bf16)h;
  hcur_bf[idx]  = (__bf16)h;
  hcur_f[idx]   = h;
  if (hs_bf) hs_bf[idx] = (__bf16)h;
}

__global__ __launch_bounds__(256)
void mask_kernel(const int* __restrict__ tok, float* __restrict__ out)
{
  int idx = blockIdx.x * 256 + threadIdx.x;
  if (idx < T_DIM * B_DIM) out[idx] = (tok[idx] != 0) ? 1.0f : 0.0f;
}

// ---------------------------------------------------------------------------
extern "C" void kernel_launch(void* const* d_in, const int* in_sizes, int n_in,
                              void* d_out, int out_size, void* d_ws, size_t ws_size,
                              hipStream_t stream)
{
  const int*   tokens  = (const int*)  d_in[0];
  const float* emb_tab = (const float*)d_in[1];
  const float* conv1_w = (const float*)d_in[2];
  const float* conv1_b = (const float*)d_in[3];
  const float* bn1_g   = (const float*)d_in[4];
  const float* bn1_b   = (const float*)d_in[5];
  const float* conv2_w = (const float*)d_in[6];
  const float* conv2_b = (const float*)d_in[7];
  const float* r_wih   = (const float*)d_in[8];
  const float* r_whh   = (const float*)d_in[9];
  const float* r_bih   = (const float*)d_in[10];
  const float* r_bhh   = (const float*)d_in[11];
  const float* r_sw    = (const float*)d_in[12];
  const float* r_sb    = (const float*)d_in[13];
  const float* p_sw    = (const float*)d_in[14];
  const float* p_sb    = (const float*)d_in[15];
  const float* ffd_w   = (const float*)d_in[16];
  const float* ffd_b   = (const float*)d_in[17];
  const float* ffd_g   = (const float*)d_in[18];
  const float* ffd_be  = (const float*)d_in[19];

  char* p = (char*)d_ws;
  auto alloc = [&](size_t bytes) -> char* {
    char* q = p; p += (bytes + 255) & ~(size_t)255; return q;
  };
  const size_t TBH = (size_t)T_DIM * B_DIM * H_DIM;

  __bf16* emb_bf    = (__bf16*)alloc(TBH * 2);
  __bf16* h1_bf     = (__bf16*)alloc(TBH * 2);
  __bf16* hs_bf     = (__bf16*)alloc(TBH * 2);
  __bf16* sels_bf   = (__bf16*)alloc(TBH * 2);
  __bf16* w1p       = (__bf16*)alloc((size_t)H_DIM * KTOT * 2);
  __bf16* sw_bf     = (__bf16*)alloc((size_t)NLAY * H_DIM * KTOT * 2);
  __bf16* pw_bf     = (__bf16*)alloc((size_t)H_DIM * KTOT * 2);
  __bf16* fw_bf     = (__bf16*)alloc((size_t)H_DIM * KTOT * 2);
  __bf16* wpack     = (__bf16*)alloc((size_t)NLAY * 4 * H_DIM * KTOT * 2);
  float*  bpack     = (float*) alloc((size_t)NLAY * 4 * H_DIM * 4);
  float*  gate      = (float*) alloc((size_t)B_DIM * T_DIM * 4);
  float*  gaten     = (float*) alloc((size_t)B_DIM * T_DIM * 4);
  float*  mg        = (float*) alloc((size_t)B_DIM * T_DIM * NSLOT * 4);
  float*  mgn       = (float*) alloc((size_t)B_DIM * T_DIM * NSLOT * 4);
  float*  rh        = (float*) alloc((size_t)NLAY * B_DIM * NSLOT * H_DIM * 4);
  float*  rc        = (float*) alloc((size_t)NLAY * B_DIM * NSLOT * H_DIM * 4);
  float*  pm        = (float*) alloc((size_t)B_DIM * NSLOT * H_DIM * 4);
  __bf16* hlast_bf  = (__bf16*)alloc((size_t)NLAY * B_DIM * H_DIM * 2);
  __bf16* pmlast_bf = (__bf16*)alloc((size_t)B_DIM * H_DIM * 2);
  __bf16* hcur_bf   = (__bf16*)alloc((size_t)B_DIM * H_DIM * 2);
  float*  hcur_f    = (float*) alloc((size_t)B_DIM * H_DIM * 4);
  __bf16* selh_bf   = (__bf16*)alloc((size_t)B_DIM * H_DIM * 2);
  float*  selc      = (float*) alloc((size_t)B_DIM * H_DIM * 4);
  float*  keybuf    = (float*) alloc((size_t)B_DIM * H_DIM * 4);
  float*  gatesbuf  = (float*) alloc((size_t)B_DIM * 4 * H_DIM * 4);
  __bf16* zbuf      = (__bf16*)alloc(1024 * 2);

  // deterministic state init (every call)
  (void)hipMemsetAsync(rh, 0, (size_t)NLAY * B_DIM * NSLOT * H_DIM * 4, stream);
  (void)hipMemsetAsync(rc, 0, (size_t)NLAY * B_DIM * NSLOT * H_DIM * 4, stream);
  (void)hipMemsetAsync(pm, 0, (size_t)B_DIM * NSLOT * H_DIM * 4, stream);
  (void)hipMemsetAsync(hlast_bf, 0, (size_t)NLAY * B_DIM * H_DIM * 2, stream);
  (void)hipMemsetAsync(pmlast_bf, 0, (size_t)B_DIM * H_DIM * 2, stream);
  (void)hipMemsetAsync(zbuf, 0, 1024 * 2, stream);

  // weight packing / conversion
  embed_kernel<<<(int)((TBH + 255) / 256), 256, 0, stream>>>(tokens, emb_tab, emb_bf);
  long nsw = (long)NLAY * H_DIM * KTOT;
  cvt_bf16_kernel<<<(int)((nsw + 255) / 256), 256, 0, stream>>>(r_sw, sw_bf, nsw);
  long nw1 = (long)H_DIM * KTOT;
  cvt_bf16_kernel<<<(int)((nw1 + 255) / 256), 256, 0, stream>>>(p_sw, pw_bf, nw1);
  cvt_bf16_kernel<<<(int)((nw1 + 255) / 256), 256, 0, stream>>>(ffd_w, fw_bf, nw1);
  pack_conv1_kernel<<<(int)(((long)H_DIM * H_DIM + 255) / 256), 256, 0, stream>>>(conv1_w, w1p);
  pack_gates_kernel<<<(int)(((long)NLAY * 4 * H_DIM * H_DIM + 255) / 256), 256, 0, stream>>>(
      r_wih, r_whh, wpack);
  pack_bias_kernel<<<(NLAY * 4 * H_DIM + 255) / 256, 256, 0, stream>>>(r_bih, r_bhh, bpack);

  // parser: conv1 GEMM (A0 row r reads emb row r-B; rows < B read zeros)
  {
    dim3 grid(H_DIM / 128, (T_DIM * B_DIM) / 16);
    gemm_bf16_wmma<<<grid, 256, 0, stream>>>(
        emb_bf - (ptrdiff_t)B_DIM * H_DIM, emb_bf, w1p,
        conv1_b, bn1_g, bn1_b, nullptr, h1_bf, zbuf,
        H_DIM, B_DIM, 1);
  }
  conv2_kernel<<<T_DIM * B_DIM, 256, 0, stream>>>(h1_bf, conv2_w, conv2_b, gate, gaten);
  mg_kernel<<<(B_DIM * T_DIM + 255) / 256, 256, 0, stream>>>(gate, gaten, mg, mgn);

  // recurrent scan
  for (int t = 0; t < T_DIM; ++t) {
    for (int j = 0; j < NLAY; ++j) {
      const __bf16* A0 = (j == 0) ? (emb_bf + (size_t)t * B_DIM * H_DIM) : hcur_bf;
      float* rhj = rh + (size_t)j * B_DIM * NSLOT * H_DIM;
      float* rcj = rc + (size_t)j * B_DIM * NSLOT * H_DIM;

      dim3 gk(H_DIM / 128, B_DIM / 16);
      gemm_bf16_wmma<<<gk, 256, 0, stream>>>(
          A0, hlast_bf + (size_t)j * B_DIM * H_DIM,
          sw_bf + (size_t)j * H_DIM * KTOT, r_sb + (size_t)j * H_DIM,
          nullptr, nullptr, keybuf, nullptr, zbuf, H_DIM, 0, 0);

      attn_kernel<<<B_DIM, 128, 0, stream>>>(rhj, rcj, keybuf, mg, t,
                                             selh_bf, selc, nullptr, nullptr);

      dim3 gg(4 * H_DIM / 128, B_DIM / 16);
      gemm_bf16_wmma<<<gg, 256, 0, stream>>>(
          A0, selh_bf, wpack + (size_t)j * 4 * H_DIM * KTOT,
          bpack + (size_t)j * 4 * H_DIM,
          nullptr, nullptr, gatesbuf, nullptr, zbuf, 4 * H_DIM, 0, 0);

      lstm_kernel<<<(B_DIM * H_DIM) / 256, 256, 0, stream>>>(
          gatesbuf, selc, rhj, rcj, t,
          hlast_bf + (size_t)j * B_DIM * H_DIM, hcur_bf, hcur_f,
          (j == NLAY - 1) ? (hs_bf + (size_t)t * B_DIM * H_DIM) : nullptr);
    }
    // predictor
    dim3 gk(H_DIM / 128, B_DIM / 16);
    gemm_bf16_wmma<<<gk, 256, 0, stream>>>(
        hcur_bf, pmlast_bf, pw_bf, p_sb,
        nullptr, nullptr, keybuf, nullptr, zbuf, H_DIM, 0, 0);
    attn_kernel<<<B_DIM, 128, 0, stream>>>(pm, nullptr, keybuf, mgn, t,
                                           sels_bf + (size_t)t * B_DIM * H_DIM,
                                           nullptr, hcur_f, pmlast_bf);
  }

  // final FFD GEMM with tanh/affine epilogue -> d_out
  {
    dim3 grid(H_DIM / 128, (T_DIM * B_DIM) / 16);
    gemm_bf16_wmma<<<grid, 256, 0, stream>>>(
        hs_bf, sels_bf, fw_bf, ffd_b, ffd_g, ffd_be,
        (float*)d_out, nullptr, zbuf, H_DIM, 0, 2);
  }
  mask_kernel<<<(T_DIM * B_DIM + 255) / 256, 256, 0, stream>>>(
      tokens, (float*)d_out + TBH);
}
